// KMEANS_88330297409964
// MI455X (gfx1250) — matrix operand order repaired and checked by reading, not compiled, not run on hardware
//
#include <hip/hip_runtime.h>

// B=65536 rows, G=256 features (GEMM K-dim), K=512 codes (GEMM N-dim)
#define NB 65536
#define NG 256
#define NK 512

typedef __attribute__((ext_vector_type(16))) __bf16 v16bf;
typedef __attribute__((ext_vector_type(8)))  float  v8f;

static __device__ __forceinline__ __bf16 f2bf(float f) {
  union { float f; unsigned u; } a; a.f = f;
  unsigned r = a.u + 0x7fffu + ((a.u >> 16) & 1u);   // round-to-nearest-even
  union { unsigned short s; __bf16 b; } c; c.s = (unsigned short)(r >> 16);
  return c.b;
}
static __device__ __forceinline__ float bf2f(__bf16 b) {
  union { __bf16 b; unsigned short s; } c; c.b = b;
  union { unsigned u; float f; } a; a.u = ((unsigned)c.s) << 16;
  return a.f;
}

// Raw LDS byte offset of a __shared__ object (generic -> AS(3) -> ptrtoint).
static __device__ __forceinline__ unsigned lds_off(const void* p) {
  return (unsigned)(size_t)(__attribute__((address_space(3))) const char*)p;
}
// Async global->LDS 16B copy (per lane), tracked by ASYNCcnt.
static __device__ __forceinline__ void async_ld_b128(unsigned dst_lds, const void* gsrc) {
  asm volatile("global_load_async_to_lds_b128 %0, %1, off"
               :: "v"(dst_lds), "v"((unsigned long long)(size_t)gsrc)
               : "memory");
}

// ---------------------------------------------------------------------------
// Prep kernel 1: swizzle mu (G,K) fp32 into WMMA B-fragment order, split into
// bf16 hi/lo. Contiguous-k B order used here (16-bit): per lane, element i holds
// K = i + 16*(lane>=16), column N = lane&15.  Fragment stream indexed as
// ((tile*8 + chunk)*32 + lane)*16 + i  -> each tile is a contiguous 8KB.
// ---------------------------------------------------------------------------
__global__ void build_mu_frags(const float* __restrict__ mu,
                               __bf16* __restrict__ hi, __bf16* __restrict__ lo) {
  int idx   = blockIdx.x * blockDim.x + threadIdx.x;  // 0 .. 32*8*32*16-1
  int i     = idx & 15;
  int lane  = (idx >> 4) & 31;
  int chunk = (idx >> 9) & 7;
  int tile  = idx >> 12;                               // 0..31
  int g   = chunk * 32 + i + ((lane >> 4) << 4);       // B-matrix K index
  int col = tile * 16 + (lane & 15);                   // B-matrix N index
  float v = mu[g * NK + col];
  __bf16 h = f2bf(v);
  hi[idx] = h;
  lo[idx] = f2bf(v - bf2f(h));
}

// ---------------------------------------------------------------------------
// Prep kernel 2: m2[k] = sum_g mu[g][k]^2  and  muT[k][g] = mu[g][k]
// ---------------------------------------------------------------------------
__global__ void build_m2_muT(const float* __restrict__ mu,
                             float* __restrict__ m2, float* __restrict__ muT) {
  int col = blockIdx.x * blockDim.x + threadIdx.x;     // 0..511
  float s = 0.0f;
  for (int g = 0; g < NG; ++g) {
    float v = mu[g * NK + col];
    s += v * v;
    muT[col * NG + g] = v;
  }
  m2[col] = s;
}

// ---------------------------------------------------------------------------
// Main kernel: 8 waves/block, each wave owns a 16-row strip.
//   A-fragments (images, bf16 hi/lo) resident in VGPRs.
//   B fragments staged global->LDS with async copies (ASYNCcnt), double
//   buffered two tiles deep; consumed via short-latency ds_load_b128.
//   32 col-tiles x 8 K-chunks x 3 WMMAs (bf16x3 split, fp32 accumulate),
//   three independent accumulator chains.  Argmin epilogue + coalesced
//   decode gather from muT.
// ---------------------------------------------------------------------------
__global__ __launch_bounds__(256) void kmeans_wmma(
    const float*  __restrict__ images,
    const __bf16* __restrict__ mu_hi,
    const __bf16* __restrict__ mu_lo,
    const float*  __restrict__ m2,
    const float*  __restrict__ muT,
    float*        __restrict__ out) {
  // Per-tile staging: [buf][ hi: 8KB | lo: 8KB ]
  __shared__ unsigned char smem[2][16384];
  __shared__ int lds_idx[128];

  const int tid  = threadIdx.x;
  const int lane = tid & 31;
  const int wave = tid >> 5;
  const int row0 = blockIdx.x * 128 + wave * 16;
  const int n    = lane & 15;            // A-row within strip / C-column within tile
  const int s    = (lane >> 4) * 16;     // A K-offset for upper lane half

  // Copy helper state: each thread moves 4x16B per tile (2 hi + 2 lo).
  const char* g_hi = (const char*)mu_hi;
  const char* g_lo = (const char*)mu_lo;
  const unsigned d_hi0 = lds_off(&smem[0][tid * 16]);
  const unsigned d_lo0 = d_hi0 + 8192;

  // ---- Kick off async copies for tiles 0 and 1 ----------------------------
  {
    const char* sh0 = g_hi + tid * 16;
    const char* sl0 = g_lo + tid * 16;
    async_ld_b128(d_hi0,          sh0);
    async_ld_b128(d_hi0 + 4096,   sh0 + 4096);
    async_ld_b128(d_lo0,          sl0);
    async_ld_b128(d_lo0 + 4096,   sl0 + 4096);
    const char* sh1 = sh0 + 8192;
    const char* sl1 = sl0 + 8192;
    async_ld_b128(d_hi0 + 16384,        sh1);
    async_ld_b128(d_hi0 + 16384 + 4096, sh1 + 4096);
    async_ld_b128(d_lo0 + 16384,        sl1);
    async_ld_b128(d_lo0 + 16384 + 4096, sl1 + 4096);
  }

  // ---- Load + split 16x256 image strip into resident A-fragments ----------
  // (conversion VALU work overlaps the in-flight async copies)
  // A order used here (16-bit 16x32, contiguous k): element i -> K = s + i
  v16bf a_hi[8], a_lo[8];
  const float* rb = images + (size_t)(row0 + n) * NG;
#pragma unroll
  for (int c = 0; c < 8; ++c) {
    float f[16];
    const float4* p0 = (const float4*)(rb + c * 32 + s);
    const float4* p1 = (const float4*)(rb + c * 32 + s + 8);
    float4 q;
    q = p0[0]; f[0] = q.x; f[1] = q.y; f[2]  = q.z; f[3]  = q.w;
    q = p0[1]; f[4] = q.x; f[5] = q.y; f[6]  = q.z; f[7]  = q.w;
    q = p1[0]; f[8] = q.x; f[9] = q.y; f[10] = q.z; f[11] = q.w;
    q = p1[1]; f[12] = q.x; f[13] = q.y; f[14] = q.z; f[15] = q.w;
#pragma unroll
    for (int i = 0; i < 16; ++i) {
      __bf16 h = f2bf(f[i]);
      a_hi[c][i] = h;
      a_lo[c][i] = f2bf(f[i] - bf2f(h));
    }
  }

  // ---- GEMM + running argmin over 32 column tiles --------------------------
  float bv[8];
  int   bi[8];
#pragma unroll
  for (int v = 0; v < 8; ++v) { bv[v] = 3.0e38f; bi[v] = 0; }

  for (int t = 0; t < 32; ++t) {
    // Wait for tile t's copy (tile t+1 may remain in flight: 4 instrs/wave).
    if (t < 31) asm volatile("s_wait_asynccnt 0x4" ::: "memory");
    else        asm volatile("s_wait_asynccnt 0x0" ::: "memory");
    __syncthreads();

    const __bf16* sh = (const __bf16*)smem[t & 1];          // hi fragments
    const __bf16* sl = (const __bf16*)(smem[t & 1] + 8192); // lo fragments

    v8f acc0 = {0.f, 0.f, 0.f, 0.f, 0.f, 0.f, 0.f, 0.f};
    v8f acc1 = {0.f, 0.f, 0.f, 0.f, 0.f, 0.f, 0.f, 0.f};
    v8f acc2 = {0.f, 0.f, 0.f, 0.f, 0.f, 0.f, 0.f, 0.f};
#pragma unroll
    for (int c = 0; c < 8; ++c) {
      const v16bf bh = *(const v16bf*)(sh + ((c * 32 + lane) << 4));
      const v16bf bl = *(const v16bf*)(sl + ((c * 32 + lane) << 4));
      acc0 = __builtin_amdgcn_wmma_f32_16x16x32_bf16(false, a_hi[c], false, bh,
                                                     (short)0, acc0, false, false);
      acc1 = __builtin_amdgcn_wmma_f32_16x16x32_bf16(false, a_lo[c], false, bh,
                                                     (short)0, acc1, false, false);
      acc2 = __builtin_amdgcn_wmma_f32_16x16x32_bf16(false, a_hi[c], false, bl,
                                                     (short)0, acc2, false, false);
    }

    const int   col = t * 16 + n;
    const float m2c = m2[col];
#pragma unroll
    for (int v = 0; v < 8; ++v) {
      float xm = acc0[v] + acc1[v] + acc2[v];
      float d  = m2c - 2.0f * xm;          // x^2 is row-constant: drop it
      if (d < bv[v]) { bv[v] = d; bi[v] = col; }
    }

    __syncthreads();                       // everyone done reading buf[t&1]
    if (t + 2 < 32) {                      // refill it with tile t+2
      const unsigned db = d_hi0 + (unsigned)((t & 1) * 16384);
      const char* sh2 = g_hi + (t + 2) * 8192 + tid * 16;
      const char* sl2 = g_lo + (t + 2) * 8192 + tid * 16;
      async_ld_b128(db,               sh2);
      async_ld_b128(db + 4096,        sh2 + 4096);
      async_ld_b128(db + 8192,        sl2);
      async_ld_b128(db + 8192 + 4096, sl2 + 4096);
    }
  }

  // ---- Cross-lane argmin within each 16-lane group (tie -> smaller idx) ----
#pragma unroll
  for (int off = 1; off < 16; off <<= 1) {
#pragma unroll
    for (int v = 0; v < 8; ++v) {
      float ov = __shfl_xor(bv[v], off, 32);
      int   oi = __shfl_xor(bi[v], off, 32);
      if (ov < bv[v] || (ov == bv[v] && oi < bi[v])) { bv[v] = ov; bi[v] = oi; }
    }
  }
  // C layout: lanes 0-15 hold rows v (0..7), lanes 16-31 hold rows v+8
  if (n == 0) {
    const int half = lane >> 4;
#pragma unroll
    for (int v = 0; v < 8; ++v) lds_idx[wave * 16 + half * 8 + v] = bi[v];
  }
  __syncthreads();

  // ---- Decode: recon[row,:] = muT[kbest,:], fully coalesced ---------------
#pragma unroll 1
  for (int r = 0; r < 16; ++r) {
    const int k = lds_idx[wave * 16 + r];
    const float4* src = (const float4*)(muT + (size_t)k * NG);
    float4*       dst = (float4*)(out + (size_t)(row0 + r) * NG);
    dst[lane]      = src[lane];
    dst[lane + 32] = src[lane + 32];
  }
}

// ---------------------------------------------------------------------------
extern "C" void kernel_launch(void* const* d_in, const int* in_sizes, int n_in,
                              void* d_out, int out_size, void* d_ws, size_t ws_size,
                              hipStream_t stream) {
  const float* images = (const float*)d_in[0];   // (B, G) fp32
  const float* mu     = (const float*)d_in[1];   // (G, K) fp32
  float* out = (float*)d_out;                    // (B, G) fp32

  // Workspace layout (~1.02 MB):
  //   [0KB,   264KB) mu_hi fragments (256KB data)
  //   [264KB, 528KB) mu_lo fragments (256KB data)
  //   [528KB, 530KB) m2 (512 f32)
  //   [532KB,1044KB) muT (512x256 f32)
  char* ws = (char*)d_ws;
  __bf16* mu_hi = (__bf16*)(ws);
  __bf16* mu_lo = (__bf16*)(ws + (264u << 10));
  float*  m2    = (float*)(ws + (528u << 10));
  float*  muT   = (float*)(ws + (532u << 10));

  build_mu_frags<<<512, 256, 0, stream>>>(mu, mu_hi, mu_lo);
  build_m2_muT<<<2, 256, 0, stream>>>(mu, m2, muT);
  kmeans_wmma<<<512, 256, 0, stream>>>(images, mu_hi, mu_lo, m2, muT, out);
  (void)in_sizes; (void)n_in; (void)out_size; (void)ws_size;
}
